// AdaptiveRouter_22170621182524
// MI455X (gfx1250) — matrix operand order repaired and checked by reading, not compiled, not run on hardware
//
#include <hip/hip_runtime.h>
#include <hip/hip_bf16.h>
#include <math.h>

// ---------------------------------------------------------------------------
// AdaptiveRouter for MI455X (gfx1250, wave32).
// Heavy GEMMs via v_wmma_f32_16x16x32_bf16 with 32x64 register-blocked wave
// tiles (8 WMMA per 12 b128 loads); output scatter phase is HBM-bound.
// ---------------------------------------------------------------------------

typedef __bf16 v16bf __attribute__((ext_vector_type(16)));
typedef float  v8f   __attribute__((ext_vector_type(8)));

#define Bb   2
#define Ss   2048
#define Hh   1024
#define Ee   16
#define Kk   2
#define Tt   (Bb * Ss)        // 4096 tokens
#define Nn   (Tt * Kk)        // 8192 assignments
#define CAPMAX 768

// d_out flat layout (float), in reference return order
#define OFF_DISP 0ull
#define SZ_DISP  ((size_t)Tt * Ee * CAPMAX)          // 50331648
#define OFF_COMB (OFF_DISP + SZ_DISP)
#define OFF_ROUT (OFF_COMB + SZ_DISP)                // 100663296
#define OFF_AUX  (OFF_ROUT + (size_t)Tt * Ee)        // 100728832
#define OFF_IMP  (OFF_AUX + 1)                       // 100728833

__device__ __forceinline__ unsigned short f2bf(float f) {
  unsigned u = __float_as_uint(f);
  u += 0x7fffu + ((u >> 16) & 1u);          // round-to-nearest-even
  return (unsigned short)(u >> 16);
}
__device__ __forceinline__ float bf2f(unsigned short h) {
  return __uint_as_float(((unsigned)h) << 16);
}

// ------------------------- utility kernels ---------------------------------
__global__ void zero4_k(float4* __restrict__ p, long long n4) {
  long long i = (long long)blockIdx.x * blockDim.x + threadIdx.x;
  long long stride = (long long)gridDim.x * blockDim.x;
  float4 z; z.x = z.y = z.z = z.w = 0.f;
  for (; i < n4; i += stride) p[i] = z;
}

__global__ void zero_small_k(float* __restrict__ probsum, int* __restrict__ ecnt) {
  int t = threadIdx.x;
  if (t < Ee) { probsum[t] = 0.f; ecnt[t] = 0; }
}

__global__ void cvt_k(const float* __restrict__ in, unsigned short* __restrict__ out,
                      long long n) {
  long long i = (long long)blockIdx.x * blockDim.x + threadIdx.x;
  long long stride = (long long)gridDim.x * blockDim.x;
  for (; i < n; i += stride) out[i] = f2bf(in[i]);
}

// in: [K x N] row-major fp32  ->  out: [N x K] row-major bf16 (transpose+convert)
__global__ void tcvt_k(const float* __restrict__ in, unsigned short* __restrict__ out,
                       int Kd, int Ncols) {
  long long total = (long long)Kd * Ncols;
  long long i = (long long)blockIdx.x * blockDim.x + threadIdx.x;
  long long stride = (long long)gridDim.x * blockDim.x;
  for (; i < total; i += stride) {
    int k = (int)(i / Ncols);
    int n = (int)(i % Ncols);
    out[(size_t)n * Kd + k] = f2bf(in[i]);
  }
}

// -------------------- WMMA fragment loads (bf16, wave32) -------------------
// 16-bit A 16x32 layout: lane (row = lane&15, half = lane>>4) holds
// K=[8*half..+7] in V0..3 and K=[16+8*half..+7] in V4..7 -> two b128 chunks.
__device__ __forceinline__ v16bf load_a_frag(const unsigned short* row, int k, int half) {
  v16bf a;
  ((uint4*)&a)[0] = *(const uint4*)(row + k + 8 * half);
  ((uint4*)&a)[1] = *(const uint4*)(row + k + 16 + 8 * half);
  return a;
}
// 16-bit B 32x16 layout (B stored transposed, K-contiguous): lane holds
// K=[8*half..+7] and K=[16+8*half..+7] of its column -> two b128 chunks.
__device__ __forceinline__ v16bf load_b_frag(const unsigned short* row, int k, int half) {
  v16bf b;
  ((uint4*)&b)[0] = *(const uint4*)(row + k + 8 * half);
  ((uint4*)&b)[1] = *(const uint4*)(row + k + 16 + 8 * half);
  return b;
}

__device__ __forceinline__ v8f wmma_bf16(v16bf a, v16bf b, v8f c) {
  return __builtin_amdgcn_wmma_f32_16x16x32_bf16(false, a, false, b, (short)0, c,
                                                 false, false);
}

// ------------------- register-blocked WMMA GEMM (32x64 / wave) -------------
// C[M,N] = act(A[M,K] @ Bt[N,K]^T + bias[N]); A,Bt bf16 row-major.
// 2 M-tiles x 4 N-tiles per wave: 8 v_wmma per 12 global_load_b128.
template <bool RELU, bool OUTBF>
__global__ __launch_bounds__(256)
void wmma_gemm_blk_k(const unsigned short* __restrict__ A,
                     const unsigned short* __restrict__ Bt,
                     const float* __restrict__ bias,
                     void* __restrict__ Cout,
                     int M, int N, int Kd) {
  const int wave = threadIdx.x >> 5;
  const int lane = threadIdx.x & 31;
  const int tn   = N >> 6;                          // 64-wide macro-columns
  const int tile = blockIdx.x * 8 + wave;
  if (tile >= (M >> 5) * tn) return;                // wave-uniform exit
  const int m0   = (tile / tn) << 5;
  const int n0   = (tile % tn) << 6;
  const int half = lane >> 4;
  const int r15  = lane & 15;

  const unsigned short* arow0 = A + (size_t)(m0 + r15) * Kd;
  const unsigned short* arow1 = A + (size_t)(m0 + 16 + r15) * Kd;
  const unsigned short* brow0 = Bt + (size_t)(n0 + r15) * Kd;
  const unsigned short* brow1 = Bt + (size_t)(n0 + 16 + r15) * Kd;
  const unsigned short* brow2 = Bt + (size_t)(n0 + 32 + r15) * Kd;
  const unsigned short* brow3 = Bt + (size_t)(n0 + 48 + r15) * Kd;

  v8f acc[2][4];
#pragma unroll
  for (int i = 0; i < 2; ++i)
#pragma unroll
    for (int j = 0; j < 4; ++j) acc[i][j] = (v8f){0.f, 0.f, 0.f, 0.f, 0.f, 0.f, 0.f, 0.f};

  for (int k = 0; k < Kd; k += 32) {
    v16bf a0 = load_a_frag(arow0, k, half);
    v16bf a1 = load_a_frag(arow1, k, half);
    v16bf b0 = load_b_frag(brow0, k, half);
    v16bf b1 = load_b_frag(brow1, k, half);
    v16bf b2 = load_b_frag(brow2, k, half);
    v16bf b3 = load_b_frag(brow3, k, half);
    __builtin_prefetch(arow0 + k + 128, 0, 3);      // global_prefetch_b8 (near)
    __builtin_prefetch(brow0 + k + 128, 0, 3);
    acc[0][0] = wmma_bf16(a0, b0, acc[0][0]);
    acc[0][1] = wmma_bf16(a0, b1, acc[0][1]);
    acc[0][2] = wmma_bf16(a0, b2, acc[0][2]);
    acc[0][3] = wmma_bf16(a0, b3, acc[0][3]);
    acc[1][0] = wmma_bf16(a1, b0, acc[1][0]);
    acc[1][1] = wmma_bf16(a1, b1, acc[1][1]);
    acc[1][2] = wmma_bf16(a1, b2, acc[1][2]);
    acc[1][3] = wmma_bf16(a1, b3, acc[1][3]);
  }

  // C/D layout: VGPR r -> M = r + 8*half, N = lane&15
#pragma unroll
  for (int j = 0; j < 4; ++j) {
    const int col = n0 + j * 16 + r15;
    const float bv = bias[col];
#pragma unroll
    for (int i = 0; i < 2; ++i) {
#pragma unroll
      for (int r = 0; r < 8; ++r) {
        int row = m0 + i * 16 + r + 8 * half;
        float v = acc[i][j][r] + bv;
        if (RELU) v = fmaxf(v, 0.f);
        if (OUTBF) ((unsigned short*)Cout)[(size_t)row * N + col] = f2bf(v);
        else       ((float*)Cout)[(size_t)row * N + col] = v;
      }
    }
  }
}

// ------------------- single-tile WMMA GEMM (for N=16) ----------------------
template <bool RELU, bool OUTBF>
__global__ __launch_bounds__(256)
void wmma_gemm_k(const unsigned short* __restrict__ A,
                 const unsigned short* __restrict__ Bt,
                 const float* __restrict__ bias,
                 void* __restrict__ Cout,
                 int M, int N, int Kd) {
  const int wave = threadIdx.x >> 5;
  const int lane = threadIdx.x & 31;
  const int tn   = N >> 4;
  const int tile = blockIdx.x * 8 + wave;
  if (tile >= (M >> 4) * tn) return;
  const int m0   = (tile / tn) << 4;
  const int n0   = (tile % tn) << 4;
  const int half = lane >> 4;
  const int r15  = lane & 15;

  const unsigned short* arow = A  + (size_t)(m0 + r15) * Kd;
  const unsigned short* brow = Bt + (size_t)(n0 + r15) * Kd;

  v8f acc = {0.f, 0.f, 0.f, 0.f, 0.f, 0.f, 0.f, 0.f};
  for (int k = 0; k < Kd; k += 32) {
    v16bf a = load_a_frag(arow, k, half);
    v16bf b = load_b_frag(brow, k, half);
    __builtin_prefetch(arow + k + 128, 0, 3);
    __builtin_prefetch(brow + k + 128, 0, 3);
    acc = wmma_bf16(a, b, acc);
  }

  const int col = n0 + r15;
  const float bv = bias[col];
#pragma unroll
  for (int r = 0; r < 8; ++r) {
    int row = m0 + r + 8 * half;
    float v = acc[r] + bv;
    if (RELU) v = fmaxf(v, 0.f);
    if (OUTBF) ((unsigned short*)Cout)[(size_t)row * N + col] = f2bf(v);
    else       ((float*)Cout)[(size_t)row * N + col] = v;
  }
}

// ------------------- importance = sigmoid(I1 @ Wi2 + bi2) ------------------
__global__ __launch_bounds__(256)
void importance_k(const unsigned short* __restrict__ I1, const float* __restrict__ Wi2,
                  const float* __restrict__ bi2, float* __restrict__ imp_out) {
  __shared__ float red[256];
  const int t = blockIdx.x;        // token
  const int tid = threadIdx.x;
  const unsigned short* row = I1 + (size_t)t * (Hh / 2);
  float s = bf2f(row[tid]) * Wi2[tid] + bf2f(row[tid + 256]) * Wi2[tid + 256];
  red[tid] = s;
  __syncthreads();
  for (int o = 128; o > 0; o >>= 1) {
    if (tid < o) red[tid] += red[tid + o];
    __syncthreads();
  }
  if (tid == 0) {
    float x = red[0] + bi2[0];
    imp_out[t] = 1.f / (1.f + __expf(-x));
  }
}

// ------------------- column mean of hidden_states --------------------------
__global__ void colmean_k(const float* __restrict__ X, float* __restrict__ avg) {
  int c = blockIdx.x * blockDim.x + threadIdx.x;
  if (c >= Hh) return;
  float s = 0.f;
  for (int r = 0; r < Tt; ++r) s += X[(size_t)r * Hh + c];
  avg[c] = s * (1.f / (float)Tt);
}

__global__ __launch_bounds__(256)
void mean_k(const float* __restrict__ imp, float* __restrict__ outm) {
  __shared__ float red[256];
  int tid = threadIdx.x;
  float s = 0.f;
  for (int i = tid; i < Tt; i += 256) s += imp[i];
  red[tid] = s;
  __syncthreads();
  for (int o = 128; o > 0; o >>= 1) {
    if (tid < o) red[tid] += red[tid + o];
    __syncthreads();
  }
  if (tid == 0) outm[0] = red[0] * (1.f / (float)Tt);
}

// ------------------- adaptive_k predictor (tiny MLP) -----------------------
__global__ __launch_bounds__(256)
void adaptk_k(const float* __restrict__ avg, const float* __restrict__ immean,
              const float* __restrict__ Wt1, const float* __restrict__ bt1,
              const float* __restrict__ Wt2, const float* __restrict__ bt2,
              int* __restrict__ ak_out) {
  __shared__ float h[256];
  const int j = threadIdx.x;
  float s = bt1[j];
  for (int i = 0; i < Hh; ++i) s += avg[i] * Wt1[i * 256 + j];
  s += immean[0] * Wt1[Hh * 256 + j];
  h[j] = fmaxf(s, 0.f);
  __syncthreads();
  if (j == 0) {
    float l0 = bt2[0], l1 = bt2[1];
    for (int q = 0; q < 256; ++q) { l0 += h[q] * Wt2[q * 2]; l1 += h[q] * Wt2[q * 2 + 1]; }
    ak_out[0] = (l1 > l0) ? 2 : 1;   // softmax monotone; argmax ties -> lower index
  }
}

// ------------------- softmax + top-2 + renorm ------------------------------
__global__ __launch_bounds__(256)
void softmax_top2_k(const float* __restrict__ logits, float* __restrict__ router_out,
                    float* __restrict__ probsum, const int* __restrict__ akp,
                    int* __restrict__ eflat, float* __restrict__ pflat,
                    int* __restrict__ vflat) {
  int t = blockIdx.x * blockDim.x + threadIdx.x;
  if (t >= Tt) return;
  const float* lg = logits + (size_t)t * Ee;
  float m = lg[0];
#pragma unroll
  for (int e = 1; e < Ee; ++e) m = fmaxf(m, lg[e]);
  float p[Ee];
  float s = 0.f;
#pragma unroll
  for (int e = 0; e < Ee; ++e) { p[e] = __expf(lg[e] - m); s += p[e]; }
  float inv = 1.f / s;
#pragma unroll
  for (int e = 0; e < Ee; ++e) {
    p[e] *= inv;
    router_out[(size_t)t * Ee + e] = p[e];
    atomicAdd(&probsum[e], p[e]);
  }
  int i1 = 0; float p1 = p[0];
#pragma unroll
  for (int e = 1; e < Ee; ++e) if (p[e] > p1) { p1 = p[e]; i1 = e; }
  int i2 = (i1 == 0) ? 1 : 0; float p2 = p[i2];
#pragma unroll
  for (int e = 0; e < Ee; ++e) if (e != i1 && p[e] > p2) { p2 = p[e]; i2 = e; }
  const int ak = *akp;
  const float denom = p1 + ((ak == 2) ? p2 : 0.f) + 1e-8f;
  eflat[2 * t]     = i1;  pflat[2 * t]     = p1 / denom;                     vflat[2 * t]     = 1;
  eflat[2 * t + 1] = i2;  pflat[2 * t + 1] = (ak == 2) ? p2 / denom : 0.f;   vflat[2 * t + 1] = (ak == 2) ? 1 : 0;
}

// ------------------- capacity scan + scatter (single workgroup) ------------
__global__ __launch_bounds__(256)
void scan_scatter_k(const int* __restrict__ eflat, const float* __restrict__ pflat,
                    const int* __restrict__ vflat, const float* __restrict__ importance,
                    const int* __restrict__ akp,
                    float* __restrict__ dispatch, float* __restrict__ combine,
                    int* __restrict__ ecnt) {
  __shared__ int hist[Ee * 256];
  const int t = threadIdx.x;
  const int chunk = Nn / 256;               // 32
  const int i0 = t * chunk;

  int cnt[Ee];
#pragma unroll
  for (int e = 0; e < Ee; ++e) cnt[e] = 0;
  for (int i = i0; i < i0 + chunk; ++i)
    if (vflat[i]) cnt[eflat[i]]++;
#pragma unroll
  for (int e = 0; e < Ee; ++e) hist[e * 256 + t] = cnt[e];
  __syncthreads();

  if (t < Ee) {                              // exclusive prefix per expert
    int run = 0;
    for (int j = 0; j < 256; ++j) { int c = hist[t * 256 + j]; hist[t * 256 + j] = run; run += c; }
    ecnt[t] = run;                           // oh.sum(0): all valid, capacity-independent
  }
  __syncthreads();

  int base[Ee];
#pragma unroll
  for (int e = 0; e < Ee; ++e) base[e] = hist[e * 256 + t];
  const int ak = *akp;
  const int cap = 384 * ak;                  // floor(B*S*1.5*ak/E)

  for (int i = i0; i < i0 + chunk; ++i) {
    if (!vflat[i]) continue;
    int e = eflat[i];
    int pos = base[e]++;
    if (pos < cap) {
      int tok = i >> 1;                      // K=2
      float f = (importance[tok] > 0.5f) ? 2.f : 1.f;   // (1 + iw) fused
      size_t off = ((size_t)tok * Ee + e) * CAPMAX + pos;
      dispatch[off] = 1.f;
      combine[off]  = pflat[i] * f;
    }
  }
}

// ------------------- aux loss ----------------------------------------------
__global__ void aux_k(const float* __restrict__ probsum, const int* __restrict__ ecnt,
                      const int* __restrict__ akp, float* __restrict__ aux_out) {
  if (threadIdx.x == 0 && blockIdx.x == 0) {
    float ak = (float)akp[0];
    float s = 0.f;
    for (int e = 0; e < Ee; ++e)
      s += (probsum[e] / (float)Tt) * ((float)ecnt[e] / ((float)Tt * ak));
    aux_out[0] = s * (float)Ee;
  }
}

// ---------------------------------------------------------------------------
extern "C" void kernel_launch(void* const* d_in, const int* in_sizes, int n_in,
                              void* d_out, int out_size, void* d_ws, size_t ws_size,
                              hipStream_t stream) {
  const float* X   = (const float*)d_in[0];
  const float* W1  = (const float*)d_in[1];
  const float* b1  = (const float*)d_in[2];
  const float* W2  = (const float*)d_in[3];
  const float* b2  = (const float*)d_in[4];
  const float* Wi1 = (const float*)d_in[5];
  const float* bi1 = (const float*)d_in[6];
  const float* Wi2 = (const float*)d_in[7];
  const float* bi2 = (const float*)d_in[8];
  const float* Wt1 = (const float*)d_in[9];
  const float* bt1 = (const float*)d_in[10];
  const float* Wt2 = (const float*)d_in[11];
  const float* bt2 = (const float*)d_in[12];

  float* out       = (float*)d_out;
  float* dispatch  = out + OFF_DISP;
  float* combine   = out + OFF_COMB;
  float* router    = out + OFF_ROUT;
  float* aux       = out + OFF_AUX;
  float* imp       = out + OFF_IMP;

  // workspace carve (256B aligned slabs)
  char* wp = (char*)d_ws;
  auto carve = [&](size_t bytes) -> void* {
    void* r = (void*)wp;
    wp += (bytes + 255) & ~(size_t)255;
    return r;
  };
  unsigned short* Xb    = (unsigned short*)carve((size_t)Tt * Hh * 2);       // bf16 X
  unsigned short* W1t   = (unsigned short*)carve((size_t)Hh * Hh * 2);       // bf16 W1^T
  unsigned short* Wi1t  = (unsigned short*)carve((size_t)(Hh / 2) * Hh * 2); // bf16 Wi1^T
  unsigned short* W2t   = (unsigned short*)carve((size_t)Ee * Hh * 2);       // bf16 W2^T
  unsigned short* C1b   = (unsigned short*)carve((size_t)Tt * Hh * 2);       // relu(X@W1+b1)
  unsigned short* I1b   = (unsigned short*)carve((size_t)Tt * (Hh / 2) * 2); // relu(X@Wi1+bi1)
  float*          lgts  = (float*)carve((size_t)Tt * Ee * 4);
  float*          avgf  = (float*)carve((size_t)Hh * 4);
  float*          imean = (float*)carve(4);
  int*            akp   = (int*)carve(4);
  float*          psum  = (float*)carve((size_t)Ee * 4);
  int*            ecnt  = (int*)carve((size_t)Ee * 4);
  int*            eflat = (int*)carve((size_t)Nn * 4);
  float*          pflat = (float*)carve((size_t)Nn * 4);
  int*            vflat = (int*)carve((size_t)Nn * 4);

  // 1) zero the sparse outputs (403 MB -> HBM-bound phase) + accumulators
  zero4_k<<<4096, 256, 0, stream>>>((float4*)dispatch, (long long)(2 * SZ_DISP / 4));
  zero_small_k<<<1, 32, 0, stream>>>(psum, ecnt);

  // 2) bf16 staging: X as-is; weights transposed so WMMA fragments are contiguous
  cvt_k<<<2048, 256, 0, stream>>>(X, Xb, (long long)Tt * Hh);
  tcvt_k<<<1024, 256, 0, stream>>>(W1,  W1t,  Hh, Hh);
  tcvt_k<<<512,  256, 0, stream>>>(Wi1, Wi1t, Hh, Hh / 2);
  tcvt_k<<<64,   256, 0, stream>>>(W2,  W2t,  Hh, Ee);

  // 3) GEMMs on the matrix cores
  //    GEMM1: [4096,1024]x[1024,1024], blocked 32x64/wave -> 2048 wave-tiles
  wmma_gemm_blk_k<true, true><<<(Tt / 32) * (Hh / 64) / 8, 256, 0, stream>>>(
      Xb, W1t, b1, C1b, Tt, Hh, Hh);
  //    GEMM3: [4096,1024]x[1024,512], blocked -> 1024 wave-tiles
  wmma_gemm_blk_k<true, true><<<(Tt / 32) * ((Hh / 2) / 64) / 8, 256, 0, stream>>>(
      Xb, Wi1t, bi1, I1b, Tt, Hh / 2, Hh);
  //    GEMM2: [4096,1024]x[1024,16] -> N too narrow to block, single-tile path
  wmma_gemm_k<false, false><<<(Tt / 16) * (Ee / 16) / 8, 256, 0, stream>>>(
      C1b, W2t, b2, lgts, Tt, Ee, Hh);

  // 4) importance scores -> d_out
  importance_k<<<Tt, 256, 0, stream>>>(I1b, Wi2, bi2, imp);

  // 5) adaptive_k
  colmean_k<<<Hh / 256, 256, 0, stream>>>(X, avgf);
  mean_k<<<1, 256, 0, stream>>>(imp, imean);
  adaptk_k<<<1, 256, 0, stream>>>(avgf, imean, Wt1, bt1, Wt2, bt2, akp);

  // 6) softmax + top-2 + renorm; router_probs -> d_out
  softmax_top2_k<<<Tt / 256, 256, 0, stream>>>(lgts, router, psum, akp, eflat, pflat, vflat);

  // 7) capacity scan + scatter into dispatch/combine
  scan_scatter_k<<<1, 256, 0, stream>>>(eflat, pflat, vflat, imp, akp, dispatch, combine, ecnt);

  // 8) aux loss
  aux_k<<<1, 32, 0, stream>>>(psum, ecnt, akp, aux);
}
